// RWKVAttention_v1_84241488544514
// MI455X (gfx1250) — matrix operand order repaired and checked by reading, not compiled, not run on hardware
//
#include <hip/hip_runtime.h>
#include <hip/hip_bf16.h>

// ---------------------------------------------------------------------------
// RWKV attention block for MI455X (gfx1250, wave32, WMMA)
//   k = x@Wk^T+bk ; v = x@Wv^T+bv        (f16 WMMA GEMM, f32 accum)
//   o = chunked linear-recurrence scan    (3-phase parallel scan)
//   out = o@Wo^T+bo                       (f16 WMMA GEMM, f32 accum)
// GEMM: 128x128 block tile, 8 waves @ 32x64, double-buffered LDS fed by
// async global->LDS (ASYNCcnt) when the toolchain exposes the builtin.
// ---------------------------------------------------------------------------

typedef _Float16 hvec4  __attribute__((ext_vector_type(4)));
typedef _Float16 hvec8  __attribute__((ext_vector_type(8)));
typedef _Float16 hvec16 __attribute__((ext_vector_type(16)));
typedef float    fvec8  __attribute__((ext_vector_type(8)));
typedef int      ivec4  __attribute__((ext_vector_type(4)));

static constexpr int DIM   = 1024;
static constexpr int SEQ   = 2048;
static constexpr int BATCH = 4;
static constexpr int GM    = BATCH * SEQ;   // 8192 rows
static constexpr int GN    = DIM;           // 1024
static constexpr int GK    = DIM;           // 1024

// scan chunking
static constexpr int NCH   = BATCH * DIM;   // 4096 independent channels
static constexpr int CH    = 32;            // chunks along T
static constexpr int LCH   = SEQ / CH;      // 64 steps per chunk

// GEMM tiling
static constexpr int BM = 128, BN = 128, BK = 32;
static constexpr int LDT = 40;              // padded LDS row stride (halfs) = 80B, 16B aligned

#if defined(__gfx1250__) && __has_builtin(__builtin_amdgcn_global_load_async_to_lds_b128)
#define USE_ASYNC_LDS 1
#else
#define USE_ASYNC_LDS 0
#endif

typedef __attribute__((address_space(1))) ivec4 as1_ivec4;
typedef __attribute__((address_space(3))) ivec4 as3_ivec4;

__device__ __forceinline__ hvec16 cat8(hvec8 lo, hvec8 hi) {
    return __builtin_shufflevector(lo, hi, 0,1,2,3,4,5,6,7,8,9,10,11,12,13,14,15);
}

// stage one 16B chunk global -> LDS
__device__ __forceinline__ void stage16(const _Float16* __restrict__ g, _Float16* l) {
#if USE_ASYNC_LDS
    __builtin_amdgcn_global_load_async_to_lds_b128(
        (as1_ivec4*)g, (as3_ivec4*)l, 0, 0);
#else
    *(hvec8*)l = *(const hvec8*)g;
#endif
}

// ---------------------------------------------------------------------------
// fp32 -> fp16 conversion, 4 elements/thread (b128 load, b64 store)
// ---------------------------------------------------------------------------
__global__ void cvt_f32_f16(const float* __restrict__ src, _Float16* __restrict__ dst) {
    long i = ((long)blockIdx.x * blockDim.x + threadIdx.x) * 4;
    float4 v = *(const float4*)(src + i);
    hvec4 h;
    h[0] = (_Float16)v.x; h[1] = (_Float16)v.y;
    h[2] = (_Float16)v.z; h[3] = (_Float16)v.w;
    *(hvec4*)(dst + i) = h;
}

// ---------------------------------------------------------------------------
// C[M,N] = A[M,K](f16) * W[N,K](f16)^T + bias[N]   (f32 out)
// 256 threads = 8 waves (4M x 2N), wave tile 32x64 -> 2x4 WMMA fragments
// ---------------------------------------------------------------------------
__global__ __launch_bounds__(256)
void gemm_bias_f16(const _Float16* __restrict__ A, const _Float16* __restrict__ W,
                   const float* __restrict__ bias, float* __restrict__ C) {
    __shared__ __align__(16) _Float16 As[2][BM * LDT];
    __shared__ __align__(16) _Float16 Bs[2][BN * LDT];

    const int tid  = threadIdx.x;
    const int lane = tid & 31;
    const int wv   = tid >> 5;
    const int lrow = lane & 15;       // 0..15
    const int lh   = lane >> 4;       // half-wave select
    const int wm   = (wv >> 1) * 32;  // wave M offset in block tile (0..96)
    const int wn   = (wv & 1)  * 64;  // wave N offset in block tile (0,64)

    const long m0 = (long)blockIdx.y * BM;
    const long n0 = (long)blockIdx.x * BN;

    fvec8 acc[2][4] = {};

    // staging indices: 128 rows x 4 chunks(8h) per matrix = 512 b128; 2/thread/matrix
    const int r0 = tid >> 2;          // 0..63
    const int c0 = (tid & 3) * 8;     // 0,8,16,24 (halfs)

    // issue one k-panel's staging (4 async b128 per thread)
    auto stage = [&](int kb, int buf) {
        stage16(&A[(m0 + r0)      * GK + kb + c0], &As[buf][ r0       * LDT + c0]);
        stage16(&A[(m0 + r0 + 64) * GK + kb + c0], &As[buf][(r0 + 64) * LDT + c0]);
        stage16(&W[(n0 + r0)      * GK + kb + c0], &Bs[buf][ r0       * LDT + c0]);
        stage16(&W[(n0 + r0 + 64) * GK + kb + c0], &Bs[buf][(r0 + 64) * LDT + c0]);
    };

    stage(0, 0);
    int buf = 0;
    for (int kb = 0; kb < GK; kb += BK, buf ^= 1) {
        const bool more = (kb + BK < GK);
        if (more) stage(kb + BK, buf ^ 1);
#if USE_ASYNC_LDS
        // in-order ASYNCcnt: <=4 leaves only the just-issued batch outstanding
        if (more) asm volatile("s_wait_asynccnt 0x4" ::: "memory");
        else      asm volatile("s_wait_asynccnt 0x0" ::: "memory");
#else
        if (more) {
            __builtin_prefetch(&A[(m0 + r0) * GK + kb + BK + c0], 0, 1);
            __builtin_prefetch(&W[(n0 + r0) * GK + kb + BK + c0], 0, 1);
        }
#endif
        __syncthreads();

        // A fragments: 16x32 f16; lane row M=lrow; K chunks {lh*8.., 16+lh*8..}
        hvec16 af[2], bf[4];
#pragma unroll
        for (int mf = 0; mf < 2; ++mf) {
            const _Float16* ap = &As[buf][(wm + mf * 16 + lrow) * LDT];
            af[mf] = cat8(*(const hvec8*)(ap + lh * 8),
                          *(const hvec8*)(ap + 16 + lh * 8));
        }
        // B fragments: 32x16 f16; lane col N=lrow; K chunks {lh*8.., 16+lh*8..}
#pragma unroll
        for (int nf = 0; nf < 4; ++nf) {
            const _Float16* bp = &Bs[buf][(wn + nf * 16 + lrow) * LDT];
            bf[nf] = cat8(*(const hvec8*)(bp + lh * 8), *(const hvec8*)(bp + 16 + lh * 8));
        }
#pragma unroll
        for (int mf = 0; mf < 2; ++mf)
#pragma unroll
            for (int nf = 0; nf < 4; ++nf)
                acc[mf][nf] = __builtin_amdgcn_wmma_f32_16x16x32_f16(
                    false, af[mf], false, bf[nf], (short)0, acc[mf][nf], false, false);
        __syncthreads();   // protects buf before next iteration overwrites it
    }

    // epilogue: C/D layout -> VGPR r holds M = base + r + 8*lh, N = base + lrow
#pragma unroll
    for (int mf = 0; mf < 2; ++mf) {
#pragma unroll
        for (int nf = 0; nf < 4; ++nf) {
            long n   = n0 + wn + nf * 16 + lrow;
            float bn = bias[n];
            long mb  = m0 + wm + mf * 16 + lh * 8;
#pragma unroll
            for (int r = 0; r < 8; ++r)
                C[(mb + r) * GN + n] = acc[mf][nf][r] + bn;
        }
    }
}

// ---------------------------------------------------------------------------
// RWKV scan, 3-phase chunked linear recurrence:  s_t = d*s_{t-1} + x_t,
// x_t = (exp(k_t)*v_t, exp(k_t)),  s_{-1} = x_0 (reference's init carry).
// ---------------------------------------------------------------------------

// Phase 1: per (channel, chunk) partial sums P = sum_j d^{L-1-j} x_{cL+j}
__global__ void scan_phase1(const float* __restrict__ kb, const float* __restrict__ vb,
                            const float* __restrict__ w,
                            float* __restrict__ Pa, float* __restrict__ Pb) {
    int g     = blockIdx.x * blockDim.x + threadIdx.x;  // 0 .. CH*NCH-1
    int chunk = g / NCH;
    int nch   = g - chunk * NCH;
    int n     = nch >> 10;
    int c     = nch & 1023;
    float d   = __expf(-w[c]);
    const float* kp = kb + ((long)n * SEQ + (long)chunk * LCH) * DIM + c;
    const float* vp = vb + ((long)n * SEQ + (long)chunk * LCH) * DIM + c;
    float pa = 0.f, pb = 0.f;
    for (int j = 0; j < LCH; ++j) {
        float kt = kp[(long)j * DIM];
        float vt = vp[(long)j * DIM];
        float ek = __expf(kt);
        pa = d * pa + ek * vt;
        pb = d * pb + ek;
    }
    Pa[(long)chunk * NCH + nch] = pa;
    Pb[(long)chunk * NCH + nch] = pb;
}

// Phase 2: per channel, 32-step sequential combine of chunk summaries.
// E_0 = x_0 ; E_c = d^L * E_{c-1} + P_{c-1}
__global__ void scan_phase2(const float* __restrict__ kb, const float* __restrict__ vb,
                            const float* __restrict__ w,
                            const float* __restrict__ Pa, const float* __restrict__ Pb,
                            float* __restrict__ Ea, float* __restrict__ Eb) {
    int nch = blockIdx.x * blockDim.x + threadIdx.x;  // 0..NCH-1
    int n   = nch >> 10;
    int c   = nch & 1023;
    float dL = __expf(-w[c] * (float)LCH);
    float k0 = kb[(long)n * SEQ * DIM + c];
    float v0 = vb[(long)n * SEQ * DIM + c];
    float ek0 = __expf(k0);
    float ea = ek0 * v0, eb = ek0;
    Ea[nch] = ea; Eb[nch] = eb;
    for (int cc = 1; cc < CH; ++cc) {
        ea = dL * ea + Pa[(long)(cc - 1) * NCH + nch];
        eb = dL * eb + Pb[(long)(cc - 1) * NCH + nch];
        Ea[(long)cc * NCH + nch] = ea;
        Eb[(long)cc * NCH + nch] = eb;
    }
}

// Phase 3: replay each chunk from its entry state, emit outputs as f16.
__global__ void scan_phase3(const float* __restrict__ kb, const float* __restrict__ vb,
                            const float* __restrict__ w, const float* __restrict__ u,
                            const float* __restrict__ Ea, const float* __restrict__ Eb,
                            _Float16* __restrict__ oh) {
    int g     = blockIdx.x * blockDim.x + threadIdx.x;
    int chunk = g / NCH;
    int nch   = g - chunk * NCH;
    int n     = nch >> 10;
    int c     = nch & 1023;
    float d  = __expf(-w[c]);
    float uu = u[c];
    float a  = Ea[(long)chunk * NCH + nch];
    float b  = Eb[(long)chunk * NCH + nch];
    const float* kp = kb + ((long)n * SEQ + (long)chunk * LCH) * DIM + c;
    const float* vp = vb + ((long)n * SEQ + (long)chunk * LCH) * DIM + c;
    _Float16*    op = oh + ((long)n * SEQ + (long)chunk * LCH) * DIM + c;
    for (int j = 0; j < LCH; ++j) {
        float kt  = kp[(long)j * DIM];
        float vt  = vp[(long)j * DIM];
        float ekt = __expf(kt);
        float euk = __expf(uu + kt);
        float out = (a + euk * vt) / (b + euk);
        op[(long)j * DIM] = (_Float16)out;
        a = d * a + ekt * vt;
        b = d * b + ekt;
    }
}

// ---------------------------------------------------------------------------
extern "C" void kernel_launch(void* const* d_in, const int* in_sizes, int n_in,
                              void* d_out, int out_size, void* d_ws, size_t ws_size,
                              hipStream_t stream) {
    (void)in_sizes; (void)n_in; (void)out_size; (void)ws_size;
    const float* x      = (const float*)d_in[0];
    // d_in[1] = Wq, d_in[2] = bq : dead in reference (DCE'd) -> skipped
    const float* Wk     = (const float*)d_in[3];
    const float* bk     = (const float*)d_in[4];
    const float* Wv     = (const float*)d_in[5];
    const float* bv     = (const float*)d_in[6];
    const float* w_rwkv = (const float*)d_in[7];
    const float* u_rwkv = (const float*)d_in[8];
    const float* Wo     = (const float*)d_in[9];
    const float* bo     = (const float*)d_in[10];
    float* out          = (float*)d_out;

    char* ws = (char*)d_ws;
    _Float16* Xh  = (_Float16*)ws; ws += (size_t)GM * GK * 2;   // 16 MB
    _Float16* Wkh = (_Float16*)ws; ws += (size_t)GN * GK * 2;   //  2 MB
    _Float16* Wvh = (_Float16*)ws; ws += (size_t)GN * GK * 2;   //  2 MB
    _Float16* Woh = (_Float16*)ws; ws += (size_t)GN * GK * 2;   //  2 MB
    float* kbuf   = (float*)ws;    ws += (size_t)GM * GN * 4;   // 32 MB
    float* vbuf   = (float*)ws;    ws += (size_t)GM * GN * 4;   // 32 MB
    _Float16* oh  = (_Float16*)ws; ws += (size_t)GM * GN * 2;   // 16 MB
    float* Pa     = (float*)ws;    ws += (size_t)CH * NCH * 4;  // 512 KB
    float* Pb     = (float*)ws;    ws += (size_t)CH * NCH * 4;
    float* Ea     = (float*)ws;    ws += (size_t)CH * NCH * 4;
    float* Eb     = (float*)ws;    ws += (size_t)CH * NCH * 4;

    // 1) precision downcast for WMMA operands
    cvt_f32_f16<<<GM * GK / 1024, 256, 0, stream>>>(x,  Xh);
    cvt_f32_f16<<<GN * GK / 1024, 256, 0, stream>>>(Wk, Wkh);
    cvt_f32_f16<<<GN * GK / 1024, 256, 0, stream>>>(Wv, Wvh);
    cvt_f32_f16<<<GN * GK / 1024, 256, 0, stream>>>(Wo, Woh);

    // 2) k, v projections on the WMMA pipes
    dim3 ggrid(GN / BN, GM / BM);  // (8, 64)
    gemm_bias_f16<<<ggrid, 256, 0, stream>>>(Xh, Wkh, bk, kbuf);
    gemm_bias_f16<<<ggrid, 256, 0, stream>>>(Xh, Wvh, bv, vbuf);

    // 3) chunked parallel RWKV scan (32x shorter critical path than naive)
    scan_phase1<<<CH * NCH / 256, 256, 0, stream>>>(kbuf, vbuf, w_rwkv, Pa, Pb);
    scan_phase2<<<NCH / 256,      256, 0, stream>>>(kbuf, vbuf, w_rwkv, Pa, Pb, Ea, Eb);
    scan_phase3<<<CH * NCH / 256, 256, 0, stream>>>(kbuf, vbuf, w_rwkv, u_rwkv, Ea, Eb, oh);

    // 4) output projection
    gemm_bias_f16<<<ggrid, 256, 0, stream>>>(oh, Woh, bo, out);
}
